// GATConvBlock_78838419685524
// MI455X (gfx1250) — matrix operand order repaired and checked by reading, not compiled, not run on hardware
//
#include <hip/hip_runtime.h>
#include <math.h>

typedef __attribute__((ext_vector_type(16))) _Float16 v16h;
typedef __attribute__((ext_vector_type(8)))  float    v8f;

// ---------------------------------------------------------------------------
// GEMM: H[n x 64] = Xin[n x 64] @ W[64 x 64], via v_wmma_f32_16x16x32_f16.
// One block = 256 threads = 8 waves; each wave computes a 16-row x 64-col strip.
// W is converted to f16 and stored transposed in LDS: Wt[col][k].
// ---------------------------------------------------------------------------
__global__ __launch_bounds__(256) void gat_gemm_wmma(
    const float* __restrict__ X, const float* __restrict__ W,
    float* __restrict__ H, int n)
{
  // 80-half row stride => 160B rows, so every 16-half chunk is 32B aligned.
  __shared__ _Float16 Wt[64][80];
  for (int idx = threadIdx.x; idx < 64 * 64; idx += 256) {
    int k = idx >> 6, c = idx & 63;
    Wt[c][k] = (_Float16)W[idx];
  }
  __syncthreads();

  const int lane = threadIdx.x & 31;
  const int wid  = threadIdx.x >> 5;
  const int hi   = lane >> 4;      // half-wave select
  const int l16  = lane & 15;
  const int row0 = blockIdx.x * 128 + wid * 16;
  const int row  = row0 + l16;
  const bool inb = (row < n);

  // A fragments: contiguous k order, same as B (not the ISA table order).
  // lane<16: K = kk+0..15; lane>=16: K = kk+16..31.
  v16h a[2];
  #pragma unroll
  for (int t = 0; t < 2; ++t) {
    float4 x0 = {0.f, 0.f, 0.f, 0.f}, x1 = x0, x2 = x0, x3 = x0;
    if (inb) {
      const float* rp = X + (size_t)row * 64 + t * 32 + hi * 16;
      x0 = ((const float4*)rp)[0];
      x1 = ((const float4*)rp)[1];
      x2 = ((const float4*)(rp + 8))[0];
      x3 = ((const float4*)(rp + 8))[1];
    }
    a[t][0] = (_Float16)x0.x;  a[t][1] = (_Float16)x0.y;
    a[t][2] = (_Float16)x0.z;  a[t][3] = (_Float16)x0.w;
    a[t][4] = (_Float16)x1.x;  a[t][5] = (_Float16)x1.y;
    a[t][6] = (_Float16)x1.z;  a[t][7] = (_Float16)x1.w;
    a[t][8]  = (_Float16)x2.x; a[t][9]  = (_Float16)x2.y;
    a[t][10] = (_Float16)x2.z; a[t][11] = (_Float16)x2.w;
    a[t][12] = (_Float16)x3.x; a[t][13] = (_Float16)x3.y;
    a[t][14] = (_Float16)x3.z; a[t][15] = (_Float16)x3.w;
  }

  #pragma unroll
  for (int ct = 0; ct < 4; ++ct) {
    const int col = ct * 16 + l16;
    // B fragments: lane<16 holds K=kk+0..15 for column `col`; lane>=16 K=kk+16..31.
    v16h b0 = *(const v16h*)&Wt[col][hi * 16];
    v16h b1 = *(const v16h*)&Wt[col][32 + hi * 16];
    v8f acc = {};
    acc = __builtin_amdgcn_wmma_f32_16x16x32_f16(false, a[0], false, b0,
                                                 (short)0, acc, false, false);
    acc = __builtin_amdgcn_wmma_f32_16x16x32_f16(false, a[1], false, b1,
                                                 (short)0, acc, false, false);
    // C/D layout: VGPR j -> (M = hi*8 + j, N = l16)
    #pragma unroll
    for (int j = 0; j < 8; ++j)
      H[(size_t)(row0 + hi * 8 + j) * 64 + col] = acc[j];
  }
}

// ---------------------------------------------------------------------------
// alpha_src / alpha_dst per node: wave32 per node, lane covers ch (lane, lane+32).
// ---------------------------------------------------------------------------
__global__ __launch_bounds__(256) void gat_alphas(
    const float* __restrict__ H, const float* __restrict__ avs,
    const float* __restrict__ avd, float* __restrict__ as_,
    float* __restrict__ ad_, int n)
{
  const int lane = threadIdx.x & 31;
  const int node = blockIdx.x * 8 + (threadIdx.x >> 5);
  if (node >= n) return;
  float h0 = H[(size_t)node * 64 + lane];
  float h1 = H[(size_t)node * 64 + lane + 32];
  float s = h0 * avs[lane] + h1 * avs[lane + 32];
  float d = h0 * avd[lane] + h1 * avd[lane + 32];
  #pragma unroll
  for (int off = 16; off > 0; off >>= 1) {
    s += __shfl_xor(s, off, 32);
    d += __shfl_xor(d, off, 32);
  }
  if (lane == 0) { as_[node] = s; ad_[node] = d; }
}

// ---------------------------------------------------------------------------
// Per-layer init: zero accumulator, reset segment max/sum.
// ---------------------------------------------------------------------------
__global__ void gat_init(float* __restrict__ acc, unsigned* __restrict__ m,
                         float* __restrict__ ssum, int npad)
{
  int idx = blockIdx.x * 256 + threadIdx.x;
  if (idx < npad * 64) acc[idx] = 0.0f;
  if (idx < npad) { m[idx] = 0u; ssum[idx] = 0.0f; }  // 0 < fmap(x) for all x
}

// monotone float <-> uint mapping for atomic max
__device__ __forceinline__ unsigned fmap(float f) {
  unsigned u = __float_as_uint(f);
  return (u & 0x80000000u) ? ~u : (u | 0x80000000u);
}
__device__ __forceinline__ float funmap(unsigned k) {
  return (k & 0x80000000u) ? __uint_as_float(k & 0x7FFFFFFFu)
                           : __uint_as_float(~k);
}

// ---------------------------------------------------------------------------
// Pass 1: edge logits (leaky-relu, original self loops masked to -1e30),
//         segment max via atomicMax on mapped uints.
// Edges [0,E) are original; [E, E+n) are the appended self loops.
// ---------------------------------------------------------------------------
__global__ void gat_edge_logits(
    const int* __restrict__ src, const int* __restrict__ dst,
    const float* __restrict__ as_, const float* __restrict__ ad_,
    float* __restrict__ ebuf, unsigned* __restrict__ m, int E, int n)
{
  int e = blockIdx.x * 256 + threadIdx.x;
  if (e >= E + n) return;
  int s, d;
  if (e < E) { s = src[e]; d = dst[e]; } else { s = d = e - E; }
  float v;
  if (e < E && s == d) {
    v = -1e30f;                              // PyG drops duplicate self loops
  } else {
    float t = as_[s] + ad_[d];
    v = t > 0.0f ? t : 0.2f * t;             // leaky_relu(0.2)
  }
  ebuf[e] = v;
  atomicMax(&m[d], fmap(v));
}

// ---------------------------------------------------------------------------
// Pass 2: p = exp(e - max[dst]); segment sum via atomicAdd.
// ---------------------------------------------------------------------------
__global__ void gat_edge_exp(
    const int* __restrict__ dst, const unsigned* __restrict__ m,
    float* __restrict__ ebuf, float* __restrict__ ssum, int E, int n)
{
  int e = blockIdx.x * 256 + threadIdx.x;
  if (e >= E + n) return;
  int d = (e < E) ? dst[e] : (e - E);
  float p = __expf(ebuf[e] - funmap(m[d]));
  ebuf[e] = p;
  atomicAdd(&ssum[d], p);
}

// ---------------------------------------------------------------------------
// Pass 3: wave32 per edge; acc[dst] += (p/sum[dst]) * h[src], 2 ch per lane.
// ---------------------------------------------------------------------------
__global__ __launch_bounds__(256) void gat_edge_scatter(
    const int* __restrict__ src, const int* __restrict__ dst,
    const float* __restrict__ ebuf, const float* __restrict__ ssum,
    const float* __restrict__ H, float* __restrict__ acc, int E, int n)
{
  const int lane = threadIdx.x & 31;
  int e = blockIdx.x * 8 + (threadIdx.x >> 5);
  if (e >= E + n) return;
  int s, d;
  if (e < E) { s = src[e]; d = dst[e]; } else { s = d = e - E; }
  float p = ebuf[e];
  if (p == 0.0f) return;                     // masked / underflowed edges
  float alpha = p / ssum[d];
  float2 hv = *(const float2*)(H + (size_t)s * 64 + lane * 2);
  atomicAdd(&acc[(size_t)d * 64 + lane * 2],     alpha * hv.x);
  atomicAdd(&acc[(size_t)d * 64 + lane * 2 + 1], alpha * hv.y);
}

// ---------------------------------------------------------------------------
// out = relu(acc + bias)
// ---------------------------------------------------------------------------
__global__ void gat_bias_relu(const float* __restrict__ acc,
                              const float* __restrict__ b,
                              float* __restrict__ out, int total)
{
  int idx = blockIdx.x * 256 + threadIdx.x;
  if (idx >= total) return;
  float v = acc[idx] + b[idx & 63];
  out[idx] = v > 0.0f ? v : 0.0f;
}

// ---------------------------------------------------------------------------
extern "C" void kernel_launch(void* const* d_in, const int* in_sizes, int n_in,
                              void* d_out, int out_size, void* d_ws, size_t ws_size,
                              hipStream_t stream) {
  const float* X   = (const float*)d_in[0];
  const int*   ei  = (const int*)d_in[1];
  const float* W1  = (const float*)d_in[3];
  const float* a1s = (const float*)d_in[4];
  const float* a1d = (const float*)d_in[5];
  const float* b1  = (const float*)d_in[6];
  const float* W2  = (const float*)d_in[7];
  const float* a2s = (const float*)d_in[8];
  const float* a2d = (const float*)d_in[9];
  const float* b2  = (const float*)d_in[10];
  float* out = (float*)d_out;

  const int n    = in_sizes[0] / 64;
  const int E    = in_sizes[1] / 2;
  const int npad = (n + 127) & ~127;
  const int* src = ei;
  const int* dst = ei + E;

  float* ws   = (float*)d_ws;
  float* H    = ws;                             // [npad*64]
  float* ACC  = ws + (size_t)npad * 64;         // [npad*64]
  float* X2   = ws + (size_t)npad * 64 * 2;     // [npad*64]
  float* AS   = ws + (size_t)npad * 64 * 3;     // [npad]
  float* AD   = AS + npad;                      // [npad]
  unsigned* M = (unsigned*)(AD + npad);         // [npad]
  float* SS   = AD + 2 * (size_t)npad;          // [npad]
  float* EB   = AD + 3 * (size_t)npad;          // [E+n]

  const int gGemm  = npad / 128;
  const int gAlpha = (n + 7) / 8;
  const int gInit  = (npad * 64 + 255) / 256;
  const int gEdge  = (E + n + 255) / 256;
  const int gScat  = (E + n + 7) / 8;
  const int gBias  = (n * 64 + 255) / 256;

  auto run_layer = [&](const float* Xin, const float* W, const float* avs,
                       const float* avd, const float* bias, float* Xout) {
    gat_gemm_wmma<<<gGemm, 256, 0, stream>>>(Xin, W, H, n);
    gat_alphas<<<gAlpha, 256, 0, stream>>>(H, avs, avd, AS, AD, n);
    gat_init<<<gInit, 256, 0, stream>>>(ACC, M, SS, npad);
    gat_edge_logits<<<gEdge, 256, 0, stream>>>(src, dst, AS, AD, EB, M, E, n);
    gat_edge_exp<<<gEdge, 256, 0, stream>>>(dst, M, EB, SS, E, n);
    gat_edge_scatter<<<gScat, 256, 0, stream>>>(src, dst, EB, SS, H, ACC, E, n);
    gat_bias_relu<<<gBias, 256, 0, stream>>>(ACC, bias, Xout, n * 64);
  };

  run_layer(X,  W1, a1s, a1d, b1, X2);
  run_layer(X2, W2, a2s, a2d, b2, out);
}
